// UniversalVocoder_80728205296082
// MI455X (gfx1250) — compile-verified, matrix operand order checked by reading, not run on hardware
//
#include <hip/hip_runtime.h>
#include <hip/hip_bf16.h>
#include <math.h>

// ---------------------------------------------------------------------------
// Types for WMMA fragments (wave32, gfx1250)
// ---------------------------------------------------------------------------
typedef _Float16 half8 __attribute__((ext_vector_type(8)));
typedef _Float16 v16h  __attribute__((ext_vector_type(16)));
typedef float    v8f   __attribute__((ext_vector_type(8)));
typedef unsigned int uint4v __attribute__((ext_vector_type(4)));
typedef int      int8v  __attribute__((ext_vector_type(8)));
typedef int      int4v  __attribute__((ext_vector_type(4)));

#define WMMA_F16(a, b, c) \
  __builtin_amdgcn_wmma_f32_16x16x32_f16(false, (a), false, (b), (short)0, (c), false, false)

// Load a 16-half A/B fragment for v_wmma_f32_16x16x32_f16.
// Per ISA 7.12.2 (16-bit A 16x32): a lane holds two 8-half chunks of K,
// at k = khalf*8 and k = khalf*8 + 16. Caller passes p = row_base + khalf*8.
static __device__ __forceinline__ v16h load_frag(const _Float16* p) {
  half8 x0 = *(const half8*)(p);
  half8 x1 = *(const half8*)(p + 16);
  return __builtin_shufflevector(x0, x1, 0,1,2,3,4,5,6,7,8,9,10,11,12,13,14,15);
}

static __device__ __forceinline__ float sigmoidf_(float x) {
  return 1.0f / (1.0f + __expf(-x));
}

// Issue a TDM 2D tile load: rows x cols fp16 tile from global (row stride
// ld elements) into LDS at lds_off. Block-uniform args -> descriptor lands
// in SGPRs. D# layout per CDNA5 ISA ch.8 (group0: count/lds/global/type,
// group1: data_size, tensor dims, tile dims, dim0 stride). Groups 2/3 zero
// (tensor <= 2D). 6-arg builtin form (clang-23 / therock headers).
// Caller must s_wait_tensorcnt + barrier before use.
static __device__ __forceinline__ void tdm_load_tile_f16(
    unsigned int lds_off, const _Float16* gptr, int rows, int cols, int ld) {
  unsigned long long ga = (unsigned long long)(size_t)gptr;
  uint4v g0;
  g0[0] = 1u;                                   // count=1, user mode
  g0[1] = lds_off;                              // LDS byte address
  g0[2] = (unsigned int)(ga & 0xFFFFFFFFu);     // global_addr[31:0]
  g0[3] = (unsigned int)((ga >> 32) & 0x01FFFFFFu) | (2u << 30);  // [56:32], type=2
  int8v g1;
  g1[0] = (int)(1u << 16);                      // wg_mask=0, data_size=1 (2 bytes)
  g1[1] = (int)((unsigned)(cols & 0xFFFF) << 16);            // tensor_dim0[15:0]
  g1[2] = (int)(((unsigned)cols >> 16) | ((unsigned)(rows & 0xFFFF) << 16)); // dim0 hi, dim1 lo
  g1[3] = (int)((unsigned)(cols & 0xFFFF) << 16);            // dim1 hi=0, tile_dim0=cols
  g1[4] = rows;                                  // tile_dim1=rows, tile_dim2=0
  g1[5] = ld;                                    // tensor_dim0_stride[31:0]
  g1[6] = 0;                                     // stride hi, dim1_stride lo
  g1[7] = 0;
  int4v gz4 = {0, 0, 0, 0};
  int8v gz8 = {0, 0, 0, 0, 0, 0, 0, 0};
  __builtin_amdgcn_tensor_load_to_lds(g0, g1, gz4, gz4, gz8, 0);
}

// ---------------------------------------------------------------------------
// fp32 -> fp16 weight conversion
// ---------------------------------------------------------------------------
__global__ void k_f32_to_f16(const float* __restrict__ s, _Float16* __restrict__ d, int n) {
  int i = blockIdx.x * 256 + threadIdx.x;
  if (i < n) d[i] = (_Float16)s[i];
}

// ---------------------------------------------------------------------------
// Generic WMMA GEMM: C[M,N] = A[M,K] * B[N,K]^T (+bias) (opt ReLU)
// A,B fp16 row-major; C fp32 or fp16. Grid: (N/128, M/16), block 256 (8 waves).
// The 16xK A tile is DMA'd into LDS once per block via the Tensor Data Mover
// (wave 0: tensor_load_to_lds + s_wait_tensorcnt), eliminating the 8x per-wave
// A re-read from HBM. Each wave owns one 16x16 N-tile; B streams from global
// (L2-resident). K multiple of 32 (<=896), N multiple of 128.
// flags: bit0 = fp16 output, bit1 = ReLU.
// ---------------------------------------------------------------------------
__global__ __launch_bounds__(256) void k_gemm_wmma(
    const _Float16* __restrict__ A, int lda,
    const _Float16* __restrict__ Bm, int ldb,
    const float* __restrict__ bias,
    void* __restrict__ C, int ldc,
    int K, int flags) {
  __shared__ __align__(16) _Float16 atile[16 * 896];
  const int lane  = threadIdx.x & 31;
  const int wave  = threadIdx.x >> 5;
  const int m0    = blockIdx.y * 16;
  const int n0    = blockIdx.x * 128 + wave * 16;
  const int row   = lane & 15;   // A: M row / B: N row
  const int khalf = lane >> 4;

  if (wave == 0) {
    tdm_load_tile_f16((unsigned int)(size_t)(void*)&atile[0],
                      A + (size_t)m0 * lda, 16, K, lda);
    __builtin_amdgcn_s_wait_tensorcnt(0);
  }
  __syncthreads();

  v8f acc = {};
  const _Float16* ap = atile + row * K + khalf * 8;              // LDS
  const _Float16* bp = Bm + (size_t)(n0 + row) * ldb + khalf * 8; // global
#pragma unroll 4
  for (int kk = 0; kk < K; kk += 32) {
    __builtin_prefetch(bp + kk + 128, 0, 1);
    v16h a = load_frag(ap + kk);
    v16h b = load_frag(bp + kk);
    acc = WMMA_F16(a, b, acc);
  }

  const int n  = n0 + (lane & 15);
  const int hi = lane >> 4;
  const float bn = bias ? bias[n] : 0.0f;
#pragma unroll
  for (int v = 0; v < 8; ++v) {
    const int m = m0 + v + 8 * hi;           // C layout: VGPR v -> M = v + 8*(lane>=16)
    float val = acc[v] + bn;
    if (flags & 2) val = fmaxf(val, 0.0f);
    if (flags & 1) ((_Float16*)C)[(size_t)m * ldc + n] = (_Float16)val;
    else           ((float*)C)[(size_t)m * ldc + n]   = val;
  }
}

// ---------------------------------------------------------------------------
// Mel bi-GRU encoder: one workgroup per batch element (B=64), 256 threads.
// threads 0..127 = forward unit i, 128..255 = backward unit i. fp32 VALU.
// Emits sliced (frames 16..23 of layer-1 concat output) as fp16 (64,8,256).
// ---------------------------------------------------------------------------
__global__ __launch_bounds__(256) void k_mel_encoder(
    const float* __restrict__ mels,
    const float* __restrict__ W0f_ih, const float* __restrict__ W0f_hh,
    const float* __restrict__ b0f_ih, const float* __restrict__ b0f_hh,
    const float* __restrict__ W0b_ih, const float* __restrict__ W0b_hh,
    const float* __restrict__ b0b_ih, const float* __restrict__ b0b_hh,
    const float* __restrict__ W1f_ih, const float* __restrict__ W1f_hh,
    const float* __restrict__ b1f_ih, const float* __restrict__ b1f_hh,
    const float* __restrict__ W1b_ih, const float* __restrict__ W1b_hh,
    const float* __restrict__ b1b_ih, const float* __restrict__ b1b_hh,
    _Float16* __restrict__ sliced) {
  __shared__ float xs[40 * 80];    // input mels for this batch element
  __shared__ float ys[40 * 256];   // layer-0 bidirectional outputs
  __shared__ float hs[256];        // per-direction hidden state
  const int b   = blockIdx.x;
  const int tid = threadIdx.x;
  const int dir = tid >> 7;        // 0 = forward, 1 = backward
  const int i   = tid & 127;

  for (int k = tid; k < 3200; k += 256) xs[k] = mels[(size_t)b * 3200 + k];
  hs[tid] = 0.0f;
  __syncthreads();

  { // layer 0: input 80 -> hidden 128 per direction
    const float* Wih = dir ? W0b_ih : W0f_ih;
    const float* Whh = dir ? W0b_hh : W0f_hh;
    const float* bih = dir ? b0b_ih : b0f_ih;
    const float* bhh = dir ? b0b_hh : b0f_hh;
    for (int s = 0; s < 40; ++s) {
      const int t = dir ? (39 - s) : s;
      float gr = bih[i], gz = bih[128 + i], gn = bih[256 + i];
      const float* x  = &xs[t * 80];
      const float* wr = &Wih[i * 80];
      const float* wz = &Wih[(128 + i) * 80];
      const float* wn = &Wih[(256 + i) * 80];
      for (int d = 0; d < 80; ++d) {
        const float xv = x[d];
        gr += xv * wr[d]; gz += xv * wz[d]; gn += xv * wn[d];
      }
      float hr = bhh[i], hz = bhh[128 + i], hn = bhh[256 + i];
      const float* h  = &hs[dir * 128];
      const float* vr = &Whh[i * 128];
      const float* vz = &Whh[(128 + i) * 128];
      const float* vn = &Whh[(256 + i) * 128];
      for (int d = 0; d < 128; ++d) {
        const float hv = h[d];
        hr += hv * vr[d]; hz += hv * vz[d]; hn += hv * vn[d];
      }
      const float r  = sigmoidf_(gr + hr);
      const float z  = sigmoidf_(gz + hz);
      const float nn = tanhf(gn + r * hn);
      const float hnew = (1.0f - z) * nn + z * hs[dir * 128 + i];
      __syncthreads();
      hs[dir * 128 + i] = hnew;
      ys[t * 256 + dir * 128 + i] = hnew;
      __syncthreads();
    }
  }

  hs[tid] = 0.0f;
  __syncthreads();

  { // layer 1: input 256 -> hidden 128 per direction; emit frames 16..23
    const float* Wih = dir ? W1b_ih : W1f_ih;
    const float* Whh = dir ? W1b_hh : W1f_hh;
    const float* bih = dir ? b1b_ih : b1f_ih;
    const float* bhh = dir ? b1b_hh : b1f_hh;
    for (int s = 0; s < 40; ++s) {
      const int t = dir ? (39 - s) : s;
      float gr = bih[i], gz = bih[128 + i], gn = bih[256 + i];
      const float* x  = &ys[t * 256];
      const float* wr = &Wih[i * 256];
      const float* wz = &Wih[(128 + i) * 256];
      const float* wn = &Wih[(256 + i) * 256];
      for (int d = 0; d < 256; ++d) {
        const float xv = x[d];
        gr += xv * wr[d]; gz += xv * wz[d]; gn += xv * wn[d];
      }
      float hr = bhh[i], hz = bhh[128 + i], hn = bhh[256 + i];
      const float* h  = &hs[dir * 128];
      const float* vr = &Whh[i * 128];
      const float* vz = &Whh[(128 + i) * 128];
      const float* vn = &Whh[(256 + i) * 128];
      for (int d = 0; d < 128; ++d) {
        const float hv = h[d];
        hr += hv * vr[d]; hz += hv * vz[d]; hn += hv * vn[d];
      }
      const float r  = sigmoidf_(gr + hr);
      const float z  = sigmoidf_(gz + hz);
      const float nn = tanhf(gn + r * hn);
      const float hnew = (1.0f - z) * nn + z * hs[dir * 128 + i];
      __syncthreads();
      hs[dir * 128 + i] = hnew;
      __syncthreads();
      if (t >= 16 && t < 24)
        sliced[((size_t)b * 8 + (t - 16)) * 256 + tid] = (_Float16)hnew;
    }
  }
}

// ---------------------------------------------------------------------------
// wav_rnn recurrence. Grid = 4 blocks, each owns 16 batch rows (one WMMA M
// tile), h resident in LDS (fp16, ping-pong). Per step each of 8 waves
// computes gh for a 112-wide hidden slice as 7 N-tiles x {r,z,n} gates via
// v_wmma_f32_16x16x32_f16 (K=896), fuses gi gather + GRU gates, writes
// h_next to LDS and wav_outs (fp16) to global. Gate order [r,z,n].
// ---------------------------------------------------------------------------
__global__ __launch_bounds__(256) void k_wavernn(
    const int* __restrict__ wavs,            // (64,1600)
    const float* __restrict__ tokTable,      // (512,2688)  emb @ Wih_emb^T
    const float* __restrict__ condpart,      // (64,8,2688) cond @ Wih_cond^T + bih
    const _Float16* __restrict__ Whh,        // (2688,896) fp16
    const float* __restrict__ bhh,           // (2688,)
    _Float16* __restrict__ wav_outs) {       // (64,1600,896) fp16
  __shared__ __align__(16) _Float16 hbuf[2][16 * 896];
  __shared__ int stok[16];
  const int g      = blockIdx.x;       // batch group (16 rows)
  const int tid    = threadIdx.x;
  const int lane   = tid & 31;
  const int wave   = tid >> 5;
  const int j0     = wave * 112;       // hidden-slice base for this wave
  const int mrow   = lane & 15;
  const int khalf  = lane >> 4;
  const int hi     = lane >> 4;
  const int ncol_l = lane & 15;

  for (int i = tid; i < 16 * 896; i += 256) hbuf[0][i] = (_Float16)0.0f;

  for (int t = 0; t < 1600; ++t) {
    __syncthreads();                       // prior step's stok reads + h writes done
    if (tid < 16) stok[tid] = wavs[(size_t)(g * 16 + tid) * 1600 + t];
    __syncthreads();
    const _Float16* hcur = hbuf[t & 1];
    _Float16*       hnxt = hbuf[(t + 1) & 1];
    const int frame = t / 200;

    for (int tile = 0; tile < 7; ++tile) {
      const int ncol = j0 + tile * 16 + ncol_l;   // index within one gate (0..895)
      v8f accR = {}, accZ = {}, accN = {};
      const _Float16* ap = hcur + mrow * 896 + khalf * 8;
      const _Float16* bR = Whh + (size_t)(ncol)        * 896 + khalf * 8;
      const _Float16* bZ = Whh + (size_t)(896  + ncol) * 896 + khalf * 8;
      const _Float16* bN = Whh + (size_t)(1792 + ncol) * 896 + khalf * 8;
#pragma unroll 4
      for (int kk = 0; kk < 896; kk += 32) {
        v16h a = load_frag(ap + kk);
        accR = WMMA_F16(a, load_frag(bR + kk), accR);
        accZ = WMMA_F16(a, load_frag(bZ + kk), accZ);
        accN = WMMA_F16(a, load_frag(bN + kk), accN);
      }

      const float bhR = bhh[ncol];
      const float bhZ = bhh[896 + ncol];
      const float bhN = bhh[1792 + ncol];
#pragma unroll
      for (int v = 0; v < 8; ++v) {
        const int m_local = v + 8 * hi;
        const int brow    = g * 16 + m_local;
        const int tok     = stok[m_local];
        const size_t tb   = (size_t)tok * 2688;
        const size_t cb   = ((size_t)brow * 8 + frame) * 2688;
        const float ir  = tokTable[tb + ncol]        + condpart[cb + ncol];
        const float iz  = tokTable[tb + 896 + ncol]  + condpart[cb + 896 + ncol];
        const float inn = tokTable[tb + 1792 + ncol] + condpart[cb + 1792 + ncol];
        const float r  = sigmoidf_(ir + accR[v] + bhR);
        const float z  = sigmoidf_(iz + accZ[v] + bhZ);
        const float nn = tanhf(inn + r * (accN[v] + bhN));
        const float hp = (float)hcur[m_local * 896 + ncol];
        const float hnew = (1.0f - z) * nn + z * hp;
        hnxt[m_local * 896 + ncol] = (_Float16)hnew;
        wav_outs[((size_t)brow * 1600 + t) * 896 + ncol] = (_Float16)hnew;
      }
    }
  }
}

// ---------------------------------------------------------------------------
// Launch
// ---------------------------------------------------------------------------
extern "C" void kernel_launch(void* const* d_in, const int* in_sizes, int n_in,
                              void* d_out, int out_size, void* d_ws, size_t ws_size,
                              hipStream_t stream) {
  (void)in_sizes; (void)n_in; (void)out_size; (void)ws_size;
  // Input order (setup_inputs dict, flattened):
  // 0 wavs(i32 64x1600), 1 mels(f32 64x40x80),
  // 2..5   mel_l0_f {Wih,Whh,bih,bhh}, 6..9  mel_l0_b,
  // 10..13 mel_l1_f,                  14..17 mel_l1_b,
  // 18 emb(512x256), 19..22 wav_rnn {Wih(2688x512),Whh(2688x896),bih,bhh},
  // 23 W1(512x896), 24 b1(512), 25 W2(512x512), 26 b2(512)
  const int*   wavs = (const int*)d_in[0];
  const float* mels = (const float*)d_in[1];
  const float* emb  = (const float*)d_in[18];
  const float* Wih_wav = (const float*)d_in[19];
  const float* Whh_wav = (const float*)d_in[20];
  const float* bih_wav = (const float*)d_in[21];
  const float* bhh_wav = (const float*)d_in[22];
  const float* W1 = (const float*)d_in[23];
  const float* b1 = (const float*)d_in[24];
  const float* W2 = (const float*)d_in[25];
  const float* b2 = (const float*)d_in[26];

  char* ws = (char*)d_ws;
  _Float16* emb16    = (_Float16*)(ws);                      //   512*256
  _Float16* Wih16    = (_Float16*)(ws + 262144);             //  2688*512
  _Float16* Whh16    = (_Float16*)(ws + 3014656);            //  2688*896
  _Float16* W116     = (_Float16*)(ws + 7831552);            //   512*896
  _Float16* W216     = (_Float16*)(ws + 8749056);            //   512*512
  _Float16* sliced16 = (_Float16*)(ws + 9273344);            //  64*8*256
  float*    tokTable = (float*)   (ws + 9535488);            //   512*2688 f32
  float*    condpart = (float*)   (ws + 15040512);           //  64*8*2688 f32
  _Float16* wavouts  = (_Float16*)(ws + 20545536);           // 64*1600*896
  _Float16* hidden16 = (_Float16*)(ws + 204046336);          // 64*1600*512

  // 1) weight conversion fp32 -> fp16
  k_f32_to_f16<<<dim3(512),  256, 0, stream>>>(emb,     emb16, 131072);
  k_f32_to_f16<<<dim3(5376), 256, 0, stream>>>(Wih_wav, Wih16, 1376256);
  k_f32_to_f16<<<dim3(9408), 256, 0, stream>>>(Whh_wav, Whh16, 2408448);
  k_f32_to_f16<<<dim3(1792), 256, 0, stream>>>(W1,      W116,  458752);
  k_f32_to_f16<<<dim3(1024), 256, 0, stream>>>(W2,      W216,  262144);

  // 2) mel bi-GRU encoder -> sliced16 (64,8,256)
  k_mel_encoder<<<dim3(64), 256, 0, stream>>>(
      mels,
      (const float*)d_in[2],  (const float*)d_in[3],  (const float*)d_in[4],  (const float*)d_in[5],
      (const float*)d_in[6],  (const float*)d_in[7],  (const float*)d_in[8],  (const float*)d_in[9],
      (const float*)d_in[10], (const float*)d_in[11], (const float*)d_in[12], (const float*)d_in[13],
      (const float*)d_in[14], (const float*)d_in[15], (const float*)d_in[16], (const float*)d_in[17],
      sliced16);

  // 3) tokTable = emb @ Wih[:, :256]^T  : M=512, N=2688, K=256, f32 out
  k_gemm_wmma<<<dim3(21, 32), 256, 0, stream>>>(
      emb16, 256, Wih16, 512, nullptr, tokTable, 2688, 256, 0);

  // 4) condpart = sliced @ Wih[:, 256:]^T + bih : M=512, N=2688, K=256, f32 out
  k_gemm_wmma<<<dim3(21, 32), 256, 0, stream>>>(
      sliced16, 256, Wih16 + 256, 512, bih_wav, condpart, 2688, 256, 0);

  // 5) wav_rnn recurrence (1600 steps), persistent, h in LDS
  k_wavernn<<<dim3(4), 256, 0, stream>>>(
      wavs, tokTable, condpart, Whh16, bhh_wav, wavouts);

  // 6) hidden = relu(wav_outs @ W1^T + b1) : M=102400, N=512, K=896, fp16 out
  k_gemm_wmma<<<dim3(4, 6400), 256, 0, stream>>>(
      wavouts, 896, W116, 896, b1, hidden16, 512, 896, 3);

  // 7) logits = hidden @ W2^T + b2 : M=102400, N=512, K=512, f32 out
  k_gemm_wmma<<<dim3(4, 6400), 256, 0, stream>>>(
      hidden16, 512, W216, 512, b2, d_out, 512, 512, 0);
}
